// DirectVoxGO_25829933318687
// MI455X (gfx1250) — matrix operand order repaired and checked by reading, not compiled, not run on hardware
//
#include <hip/hip_runtime.h>
#include <cstdint>

// ---------------- types for WMMA ----------------
typedef __bf16  bf16_t;
typedef bf16_t  v16bf __attribute__((ext_vector_type(16)));
typedef float   v8f   __attribute__((ext_vector_type(8)));
typedef int     v4i_t __attribute__((ext_vector_type(4)));

union AFrag { v16bf v; uint4 q[2]; uint32_t u[8]; };
union BFrag { v16bf v; uint32_t u[8]; };

static __device__ __forceinline__ v8f wmma_bf16(v16bf a, v16bf b, v8f c) {
  return __builtin_amdgcn_wmma_f32_16x16x32_bf16(false, a, false, b, (short)0, c, false, false);
}

static __device__ __forceinline__ uint16_t f2bf(float x) {
  union { float f; uint32_t u; } c; c.f = x;
  uint32_t r = c.u + 0x7FFFu + ((c.u >> 16) & 1u);   // round-nearest-even
  return (uint16_t)(r >> 16);
}
static __device__ __forceinline__ float bf2f(uint16_t s) {
  union { uint32_t u; float f; } c; c.u = ((uint32_t)s) << 16;
  return c.f;
}

#define GRID_D   160
#define GRID3    (160*160*160)      // 4,096,000
#define K0_DIM   12
#define WIDTH    128
#define ACT_SHIFT (-13.815510558f)  // log(1/(1-1e-6)-1)
#define LOG_EPS   (-23.025850930f)  // log(1e-10)

#define NWORDS_W0F 2048             // 8nt * 8v * 32lane
#define NWORDS_W1F 8192             // 4ks * 8nt * 8v * 32lane
#define NWORDS_ALL (NWORDS_W0F + NWORDS_W1F)   // 40 KB

// =====================================================================
// Kernel 1: swizzle w0[0:12,:] (K padded to 32) and w1[128,128] into
// bf16 B-fragment lane order (contiguous k, matching the A loads) for v_wmma_f32_16x16x32_bf16.
// Fragment word (nt, v, lane): n = nt*16 + lane%16 ; k = (lane<16?0:16)+2v
// =====================================================================
__global__ void prep_frags(const float* __restrict__ w0,
                           const float* __restrict__ w1,
                           uint32_t* __restrict__ w0f,
                           uint32_t* __restrict__ w1f)
{
  int idx = blockIdx.x * blockDim.x + threadIdx.x;
  if (idx < NWORDS_W0F) {
    int lane = idx & 31, t = idx >> 5;
    int v = t & 7, nt = t >> 3;
    int n = nt * 16 + (lane & 15);
    int k = ((lane >> 4) << 4) + 2 * v;
    uint16_t lo = (k     < K0_DIM) ? f2bf(w0[(k    ) * WIDTH + n]) : (uint16_t)0;
    uint16_t hi = (k + 1 < K0_DIM) ? f2bf(w0[(k + 1) * WIDTH + n]) : (uint16_t)0;
    w0f[idx] = (uint32_t)lo | ((uint32_t)hi << 16);
  } else if (idx < NWORDS_ALL) {
    int j = idx - NWORDS_W0F;
    int lane = j & 31, t = j >> 5;
    int v = t & 7, nt = (t >> 3) & 7, ks = t >> 6;
    int n = nt * 16 + (lane & 15);
    int k = ks * 32 + ((lane >> 4) << 4) + 2 * v;
    uint16_t lo = f2bf(w1[(k    ) * WIDTH + n]);
    uint16_t hi = f2bf(w1[(k + 1) * WIDTH + n]);
    w1f[j] = (uint32_t)lo | ((uint32_t)hi << 16);
  }
}

// =====================================================================
// Kernel 2: per-ray precompute  rayvec[r][n] = b0[n] + vemb(r)·w0[12:39,n]
// =====================================================================
__global__ void rayvec_kernel(const float* __restrict__ vd,
                              const float* __restrict__ w0,
                              const float* __restrict__ b0,
                              float* __restrict__ rayvec, int N)
{
  int idx = blockIdx.x * blockDim.x + threadIdx.x;
  if (idx >= N * WIDTH) return;
  int r = idx >> 7, n = idx & 127;
  float v3[3] = { vd[r * 3 + 0], vd[r * 3 + 1], vd[r * 3 + 2] };
  float sum = b0[n];
  #pragma unroll
  for (int d = 0; d < 3; ++d) sum += v3[d] * w0[(12 + d) * WIDTH + n];
  #pragma unroll
  for (int d = 0; d < 3; ++d) {
    float f = 1.0f;
    #pragma unroll
    for (int fi = 0; fi < 4; ++fi) {
      float ang = v3[d] * f;  f *= 2.0f;
      sum += sinf(ang) * w0[(12 + 3  + d * 4 + fi) * WIDTH + n];
      sum += cosf(ang) * w0[(12 + 15 + d * 4 + fi) * WIDTH + n];
    }
  }
  rayvec[idx] = sum;
}

// =====================================================================
// Kernel 3: per-point trilinear sampling (density + 12-ch k0),
// alpha / log(1-alpha).  k0 stored bf16 [M,16] (4 zero pad).
// =====================================================================
__global__ void sample_kernel(const float* __restrict__ dg,
                              const float* __restrict__ kg,
                              const float* __restrict__ pts,
                              uint16_t* __restrict__ k0buf,
                              float* __restrict__ alpha,
                              float* __restrict__ log1m, int M)
{
  int i = blockIdx.x * blockDim.x + threadIdx.x;
  if (i >= M) return;
  float t[3];
  #pragma unroll
  for (int d = 0; d < 3; ++d) {
    float v = (pts[i * 3 + d] + 1.0f) * 0.5f * (GRID_D - 1);
    t[d] = fminf(fmaxf(v, 0.0f), (float)(GRID_D - 1));
  }
  int ix = min((int)floorf(t[0]), GRID_D - 2);
  int iy = min((int)floorf(t[1]), GRID_D - 2);
  int iz = min((int)floorf(t[2]), GRID_D - 2);
  float fx = t[0] - ix, fy = t[1] - iy, fz = t[2] - iz;
  int ib = (ix * GRID_D + iy) * GRID_D + iz;
  const int OY = GRID_D, OX = GRID_D * GRID_D;
  float w000 = (1-fx)*(1-fy)*(1-fz), w001 = (1-fx)*(1-fy)*fz;
  float w010 = (1-fx)*fy*(1-fz),     w011 = (1-fx)*fy*fz;
  float w100 = fx*(1-fy)*(1-fz),     w101 = fx*(1-fy)*fz;
  float w110 = fx*fy*(1-fz),         w111 = fx*fy*fz;

  // density
  float d = dg[ib]*w000 + dg[ib+1]*w001 + dg[ib+OY]*w010 + dg[ib+OY+1]*w011
          + dg[ib+OX]*w100 + dg[ib+OX+1]*w101 + dg[ib+OX+OY]*w110 + dg[ib+OX+OY+1]*w111;
  float x  = d + ACT_SHIFT;
  float sp = (x > 20.0f) ? x : log1pf(expf(x));     // softplus
  float nl = -0.5f * sp;                            // log(1-alpha) = -sp*INTERVAL
  log1m[i] = fmaxf(nl, LOG_EPS);
  alpha[i] = 1.0f - expf(nl);

  // k0 channels
  #pragma unroll
  for (int c = 0; c < K0_DIM; ++c) {
    const float* g = kg + (size_t)c * GRID3;
    float v = g[ib]*w000 + g[ib+1]*w001 + g[ib+OY]*w010 + g[ib+OY+1]*w011
            + g[ib+OX]*w100 + g[ib+OX+1]*w101 + g[ib+OX+OY]*w110 + g[ib+OX+OY+1]*w111;
    k0buf[i * 16 + c] = f2bf(v);
  }
  #pragma unroll
  for (int c = K0_DIM; c < 16; ++c) k0buf[i * 16 + c] = 0;
}

// =====================================================================
// Kernel 4: one wave32 per ray. Segment exclusive scan of log1m (ray_id
// sorted -> global cumsum/searchsorted collapses to per-segment scan).
// Produces weights[M], alphainv_last -> d_out[3N+r], inits rgb = ainv.
// =====================================================================
__global__ void scan_kernel(const int* __restrict__ ray_id,
                            const float* __restrict__ alpha,
                            const float* __restrict__ log1m,
                            float* __restrict__ weights,
                            float* __restrict__ out, int N, int M)
{
  int r = blockIdx.x * (blockDim.x >> 5) + (threadIdx.x >> 5);
  if (r >= N) return;
  int lane = threadIdx.x & 31;

  auto lb = [&](int val) {
    int lo = 0, hi = M;
    while (lo < hi) { int mid = (lo + hi) >> 1; if (ray_id[mid] < val) lo = mid + 1; else hi = mid; }
    return lo;
  };
  int s = lb(r), e = lb(r + 1);

  float carry = 0.0f;
  for (int j = s; j < e; j += 32) {
    int idx = j + lane;
    bool act = idx < e;
    float xv = act ? log1m[idx] : 0.0f;
    float scan = xv;
    #pragma unroll
    for (int dlt = 1; dlt < 32; dlt <<= 1) {
      float y = __shfl_up(scan, dlt, 32);
      if (lane >= dlt) scan += y;
    }
    if (act) weights[idx] = expf(carry + (scan - xv)) * alpha[idx];
    carry += __shfl(scan, 31, 32);
  }
  float ainv = expf(carry);
  if (lane == 0) {
    out[3 * N + r]  = ainv;          // alphainv_last
    out[3 * r + 0]  = ainv;          // rgb_marched init (+ alphainv term)
    out[3 * r + 1]  = ainv;
    out[3 * r + 2]  = ainv;
  }
}

// =====================================================================
// Kernel 5: fused MLP. 128-thread blocks (4 waves); weights staged once
// per block into LDS (async global->LDS when available); each wave then
// runs 4 tiles of 16 points:
//   L0:  acc = wmma(k0_bf16[16x32-zero-pad], w0frag, rayvec-init)   (8 wmma)
//   L1:  acc2 = wmma over K=128 via LDS transpose                  (32 wmma)
//   L2:  128->3 VALU dot, sigmoid, weights[i]*rgb atomicAdd per ray
// =====================================================================
#define TILES_PER_WAVE 4
#define WAVES_PER_BLK  4
#define TILES_PER_BLK  (TILES_PER_WAVE * WAVES_PER_BLK)

__global__ void __launch_bounds__(128)
mlp_kernel(const uint16_t* __restrict__ k0buf,
           const uint32_t* __restrict__ wAll,    // w0f ++ w1f, NWORDS_ALL words
           const float* __restrict__ rayvec,
           const float* __restrict__ b1,
           const float* __restrict__ w2,
           const float* __restrict__ b2,
           const float* __restrict__ wts,
           const int*   __restrict__ ray_id,
           float* __restrict__ out,
           int tiles, int M)
{
  __shared__ uint32_t wlds[NWORDS_ALL];                    // 40 KB weight frags
  __shared__ uint16_t hbuf[WAVES_PER_BLK * 16 * WIDTH];    // 16 KB h staging
  const int tid  = threadIdx.x;
  const int wave = tid >> 5, lane = tid & 31;

  // ---- stage weight fragments into LDS (once per block) ----
#if __has_builtin(__builtin_amdgcn_global_load_async_to_lds_b128)
  for (int i = tid; i < NWORDS_ALL / 4; i += 128) {
    __builtin_amdgcn_global_load_async_to_lds_b128(
        (__attribute__((address_space(1))) v4i_t*)(wAll + i * 4),
        (__attribute__((address_space(3))) v4i_t*)(&wlds[i * 4]),
        0, 0);
  }
#if __has_builtin(__builtin_amdgcn_s_wait_asynccnt)
  __builtin_amdgcn_s_wait_asynccnt(0);
#else
  asm volatile("s_wait_asynccnt 0" ::: "memory");
#endif
#else
  for (int i = tid; i < NWORDS_ALL; i += 128) wlds[i] = wAll[i];
#endif
  __syncthreads();

  const uint32_t* w0l = wlds;
  const uint32_t* w1l = wlds + NWORDS_W0F;
  uint16_t* hw = hbuf + wave * (16 * WIDTH);

  const int l15 = lane & 15;
  const int hi8 = (lane >> 4) << 3;                 // 0 or 8
  const int half = lane >> 4;

  for (int it = 0; it < TILES_PER_WAVE; ++it) {
    int tile   = blockIdx.x * TILES_PER_BLK + wave * TILES_PER_WAVE + it;
    int tclamp = min(tile, tiles - 1);

    // ----- ray ids for this lane's 8 accumulator rows -----
    int rid8[8];
    #pragma unroll
    for (int v = 0; v < 8; ++v)
      rid8[v] = ray_id[min(tclamp * 16 + v + hi8, M - 1)];

    // ----- accumulators init = rayvec (vemb part of layer0 + b0) -----
    v8f acc[8];
    #pragma unroll
    for (int nt = 0; nt < 8; ++nt) {
      int n = nt * 16 + l15;
      #pragma unroll
      for (int v = 0; v < 8; ++v)
        acc[nt][v] = rayvec[rid8[v] * WIDTH + n];
    }

    // ----- layer 0: A = k0 (16x32, K>=12 zero) -----
    AFrag a0;
    {
      int rowA = min(tclamp * 16 + l15, M - 1);
      const uint4* ap = (const uint4*)(k0buf + (size_t)rowA * 16);
      a0.q[0] = half ? make_uint4(0u, 0u, 0u, 0u) : ap[0];
      a0.q[1] = half ? make_uint4(0u, 0u, 0u, 0u) : ap[1];
    }
    #pragma unroll
    for (int nt = 0; nt < 8; ++nt) {
      BFrag b;
      #pragma unroll
      for (int v = 0; v < 8; ++v) b.u[v] = w0l[(nt * 8 + v) * 32 + lane];
      acc[nt] = wmma_bf16(a0.v, b.v, acc[nt]);
    }

    // ----- relu + D(C-layout) -> A-layout transpose through LDS -----
    #pragma unroll
    for (int nt = 0; nt < 8; ++nt)
      #pragma unroll
      for (int v = 0; v < 8; ++v) {
        float xv = acc[nt][v];
        hw[(v + hi8) * WIDTH + nt * 16 + l15] = f2bf(xv > 0.0f ? xv : 0.0f);
      }
    __syncthreads();

    // ----- layer 1: K = 128 (4 k-steps) -----
    v8f acc2[8];
    #pragma unroll
    for (int nt = 0; nt < 8; ++nt) {
      float bv = b1[nt * 16 + l15];
      #pragma unroll
      for (int v = 0; v < 8; ++v) acc2[nt][v] = bv;
    }
    #pragma unroll
    for (int ks = 0; ks < 4; ++ks) {
      AFrag a1;
      const uint16_t* p = hw + l15 * WIDTH + ks * 32 + half * 16;
      a1.q[0] = *(const uint4*)(p);
      a1.q[1] = *(const uint4*)(p + 8);
      #pragma unroll
      for (int nt = 0; nt < 8; ++nt) {
        BFrag b;
        #pragma unroll
        for (int v = 0; v < 8; ++v) b.u[v] = w1l[((ks * 8 + nt) * 8 + v) * 32 + lane];
        acc2[nt] = wmma_bf16(a1.v, b.v, acc2[nt]);
      }
    }
    __syncthreads();

    // ----- relu + stage h1 in LDS -----
    #pragma unroll
    for (int nt = 0; nt < 8; ++nt)
      #pragma unroll
      for (int v = 0; v < 8; ++v) {
        float xv = acc2[nt][v];
        hw[(v + hi8) * WIDTH + nt * 16 + l15] = f2bf(xv > 0.0f ? xv : 0.0f);
      }
    __syncthreads();

    // ----- layer 2: 128 -> 3, sigmoid, weighted atomic accumulate -----
    const uint16_t* hp = hw + l15 * WIDTH + half * 64;
    float s0 = 0.f, s1 = 0.f, s2 = 0.f;
    #pragma unroll 8
    for (int k = 0; k < 64; ++k) {
      float hv = bf2f(hp[k]);
      const float* wr = w2 + (half * 64 + k) * 3;
      s0 = fmaf(hv, wr[0], s0);
      s1 = fmaf(hv, wr[1], s1);
      s2 = fmaf(hv, wr[2], s2);
    }
    s0 += __shfl_xor(s0, 16, 32);
    s1 += __shfl_xor(s1, 16, 32);
    s2 += __shfl_xor(s2, 16, 32);

    int row = tile * 16 + l15;
    if (half == 0 && tile < tiles && row < M) {
      float w  = wts[row];
      int  rid = ray_id[row];
      float r0 = w / (1.0f + expf(-(s0 + b2[0])));
      float r1 = w / (1.0f + expf(-(s1 + b2[1])));
      float r2 = w / (1.0f + expf(-(s2 + b2[2])));
      atomicAdd(&out[rid * 3 + 0], r0);
      atomicAdd(&out[rid * 3 + 1], r1);
      atomicAdd(&out[rid * 3 + 2], r2);
    }
    __syncthreads();
  }
}

// =====================================================================
extern "C" void kernel_launch(void* const* d_in, const int* in_sizes, int n_in,
                              void* d_out, int out_size, void* d_ws, size_t ws_size,
                              hipStream_t stream)
{
  const float* density = (const float*)d_in[0];
  const float* k0grid  = (const float*)d_in[1];
  const float* ray_pts = (const float*)d_in[2];
  const float* viewdir = (const float*)d_in[3];
  const float* w0      = (const float*)d_in[4];
  const float* b0      = (const float*)d_in[5];
  const float* w1      = (const float*)d_in[6];
  const float* b1      = (const float*)d_in[7];
  const float* w2      = (const float*)d_in[8];
  const float* b2      = (const float*)d_in[9];
  const int*   ray_id  = (const int*)  d_in[10];

  const int M = in_sizes[10];           // 4096*256
  const int N = in_sizes[3] / 3;        // 4096

  // workspace layout
  char* ws = (char*)d_ws;
  size_t o = 0;
  uint16_t* k0buf  = (uint16_t*)(ws + o); o += (size_t)M * 16 * 2;   // 32 MB
  float*    alpha  = (float*)   (ws + o); o += (size_t)M * 4;        //  4 MB
  float*    log1m  = (float*)   (ws + o); o += (size_t)M * 4;        //  4 MB
  float*    wts    = (float*)   (ws + o); o += (size_t)M * 4;        //  4 MB
  float*    rayvec = (float*)   (ws + o); o += (size_t)N * WIDTH * 4;//  2 MB
  uint32_t* w0f    = (uint32_t*)(ws + o); o += NWORDS_W0F * 4;       // contiguous
  uint32_t* w1f    = (uint32_t*)(ws + o); o += NWORDS_W1F * 4;       // with w0f

  float* out = (float*)d_out;

  prep_frags<<<(NWORDS_ALL + 255) / 256, 256, 0, stream>>>(w0, w1, w0f, w1f);
  rayvec_kernel<<<(N * WIDTH + 255) / 256, 256, 0, stream>>>(viewdir, w0, b0, rayvec, N);
  sample_kernel<<<(M + 255) / 256, 256, 0, stream>>>(density, k0grid, ray_pts,
                                                     k0buf, alpha, log1m, M);
  scan_kernel<<<(N + 3) / 4, 128, 0, stream>>>(ray_id, alpha, log1m, wts, out, N, M);

  int tiles  = (M + 15) / 16;
  int blocks = (tiles + TILES_PER_BLK - 1) / TILES_PER_BLK;
  mlp_kernel<<<blocks, 128, 0, stream>>>(k0buf, w0f, rayvec, b1, w2, b2,
                                         wts, ray_id, out, tiles, M);
}
